// TorchESN_37598143709528
// MI455X (gfx1250) — matrix operand order repaired and checked by reading, not compiled, not run on hardware
//
#include <hip/hip_runtime.h>

// ---------------- problem constants ----------------
#define SEQ   1000
#define IDIM  900
#define HDIM  4096
#define IPAD  928          // 29 * 32
#define KEXT  5024         // 157 * 32 : [1 | x(900) | pad(27) | s(4096) ]
#define MPAD  1008         // 63 * 16
#define NKB_HH 128         // 4096/32
#define NKB_IH 29          // 928/32
#define NKB_OUT 157        // 5024/32
#define NTILE_H 256        // 4096/16
#define NTILE_OUT 57       // 912/16
#define MTILES  63         // 1008/16
#define NBLK    64         // persistent grid (256 waves total)

using u16 = unsigned short;
using u32 = unsigned int;

typedef __attribute__((ext_vector_type(16))) __bf16 v16bf;
typedef __attribute__((ext_vector_type(8)))  float  v8f;
typedef __attribute__((ext_vector_type(4)))  u32    u32x4;

union U16V { u32x4 q[2]; v16bf v; };

__device__ __forceinline__ u16 f2bf(float f) {
    u32 u = __builtin_bit_cast(u32, f);
    u32 r = u + 0x7FFFu + ((u >> 16) & 1u);   // round-to-nearest-even
    return (u16)(r >> 16);
}

// ---------------- workspace layout (bytes) ----------------
static constexpr size_t OFF_CNT   = 0;
static constexpr size_t OFF_SBUF0 = 256;
static constexpr size_t OFF_SBUF1 = OFF_SBUF0 + (size_t)HDIM * 2;                 // 8448
static constexpr size_t OFF_EXT   = OFF_SBUF1 + (size_t)HDIM * 2;                 // 16640
static constexpr size_t OFF_WIH   = OFF_EXT  + (size_t)MPAD * KEXT * 2;          // +10128384
static constexpr size_t OFF_WHH   = OFF_WIH  + (size_t)NTILE_H * NKB_IH  * 1024; // +7602176
static constexpr size_t OFF_WOUT  = OFF_WHH  + (size_t)NTILE_H * NKB_HH * 1024;  // +33554432

// ---------------- init: counter=0, state->bf16, zero Ext pad rows ----------------
__global__ void esn_init(const float* __restrict__ state, u16* __restrict__ sbuf0,
                         u16* __restrict__ ext, u32* __restrict__ counter) {
    int idx = blockIdx.x * blockDim.x + threadIdx.x;
    if (idx == 0) *counter = 0u;
    if (idx < HDIM) sbuf0[idx] = f2bf(state[idx]);
    int j = idx - HDIM;
    if (j >= 0 && j < (MPAD - SEQ) * KEXT) ext[(size_t)SEQ * KEXT + j] = 0;
}

// ---------------- pack fp32 weights -> bf16 WMMA B-operand tiles ----------------
// layout: dst[((tile*nKb + kb)*32 + lane)*8 + v] = bf16x2 { W[n, k], W[n, k+1] }
//   n = tile*16 + (lane&15);  k = kb*32 + ((lane&16)?8:0) + 2v + (v>=4?8:0)
__device__ __forceinline__ int mapk(int k, int mode) {
    if (mode == 0) return k;
    if (k == 0)    return 0;          // bias column
    if (k <= IDIM) return k;          // x columns 1..900
    if (k < IPAD)  return -1;         // pad
    return k - (IPAD - IDIM - 1);     // s columns: k-27 -> 901..4996
}

__global__ void pack_bf16(const float* __restrict__ W, u32* __restrict__ dst,
                          int rows, int ldw, int cols, int nKb, int mode, int total) {
    int idx = blockIdx.x * blockDim.x + threadIdx.x;
    if (idx >= total) return;
    int lane = idx & 31;
    int rem  = idx >> 5;
    int kb   = rem % nKb;
    int tile = rem / nKb;
    int n     = tile * 16 + (lane & 15);
    int kbase = kb * 32 + ((lane & 16) ? 8 : 0);
    u32* o = dst + ((size_t)(tile * nKb + kb) * 32 + lane) * 8;
#pragma unroll
    for (int v = 0; v < 8; ++v) {
        float f0 = 0.0f, f1 = 0.0f;
        if (n < rows) {
            int c0 = mapk(kbase + 2 * v + (v & 4) * 2,     mode);
            int c1 = mapk(kbase + 2 * v + (v & 4) * 2 + 1, mode);
            if (c0 >= 0 && c0 < cols) f0 = W[(size_t)n * ldw + c0];
            if (c1 >= 0 && c1 < cols) f1 = W[(size_t)n * ldw + c1];
        }
        o[v] = (u32)f2bf(f0) | ((u32)f2bf(f1) << 16);
    }
}

// ---------------- persistent recurrence: s = tanh(W_ih x + W_hh s) ----------------
// 64 blocks x 128 threads = 256 waves; wave w owns rows [w*16, w*16+16).
// Mat-vec as WMMA: A rows all hold the same vector segment (broadcast), B holds
// 16 weight rows as columns -> D row 0 (lanes 0..15) = 16 dot products.
__global__ void __launch_bounds__(128) esn_recurrence(
        const float* __restrict__ inputs,
        const u32* __restrict__ whhP, const u32* __restrict__ wihP,
        u16* __restrict__ sbuf0, u16* __restrict__ sbuf1,
        u16* __restrict__ ext, float* __restrict__ statesOut,
        u32* __restrict__ counter) {
    __shared__ __align__(16) u16 s_lds[HDIM];
    __shared__ __align__(16) u16 x_lds[IPAD];

    const int tid  = threadIdx.x;
    const int lane = tid & 31;
    const int wid  = tid >> 5;
    const int tile = blockIdx.x * 4 + wid;      // 0..255
    const int n0   = tile * 16;
    const int sel  = (lane & 16) ? 8 : 0;       // A-operand K-half select

    const u32* bhh = whhP + (size_t)tile * NKB_HH * 256 + (size_t)lane * 8;
    const u32* bih = wihP + (size_t)tile * NKB_IH * 256 + (size_t)lane * 8;

    for (int t = 0; t < SEQ; ++t) {
        u16* sread  = (t & 1) ? sbuf1 : sbuf0;
        u16* swrite = (t & 1) ? sbuf0 : sbuf1;

        // stage bf16 state + converted input into LDS
        {
            const u32x4* src  = (const u32x4*)sread;
            u32x4*       dstl = (u32x4*)s_lds;
            for (int i = tid; i < HDIM / 8; i += 128) dstl[i] = src[i];
            for (int i = tid; i < IPAD; i += 128) {
                float xv = (i < IDIM) ? inputs[(size_t)t * IDIM + i] : 0.0f;
                x_lds[i] = f2bf(xv);
            }
        }
        __syncthreads();

        v8f acc = {};
        // W_hh @ s : 128 WMMAs
#pragma unroll 4
        for (int kb = 0; kb < NKB_HH; ++kb) {
            U16V a, b;
            const u16* sp = &s_lds[kb * 32 + sel];
            a.q[0] = *(const u32x4*)(sp);
            a.q[1] = *(const u32x4*)(sp + 16);
            const u32x4* bp = (const u32x4*)(bhh + (size_t)kb * 256);
            b.q[0] = bp[0];
            b.q[1] = bp[1];
            acc = __builtin_amdgcn_wmma_f32_16x16x32_bf16(
                false, a.v, false, b.v, (short)0, acc, false, false);
        }
        // W_ih @ x : 29 WMMAs
#pragma unroll 4
        for (int kb = 0; kb < NKB_IH; ++kb) {
            U16V a, b;
            const u16* xp = &x_lds[kb * 32 + sel];
            a.q[0] = *(const u32x4*)(xp);
            a.q[1] = *(const u32x4*)(xp + 16);
            const u32x4* bp = (const u32x4*)(bih + (size_t)kb * 256);
            b.q[0] = bp[0];
            b.q[1] = bp[1];
            acc = __builtin_amdgcn_wmma_f32_16x16x32_bf16(
                false, a.v, false, b.v, (short)0, acc, false, false);
        }

        // epilogue: D row 0 = lanes 0..15, VGPR0
        if (lane < 16) {
            int n = n0 + lane;
            float s = tanhf(acc[0]);
            statesOut[(size_t)t * HDIM + n] = s;
            u16 sb = f2bf(s);
            swrite[n] = sb;
            ext[(size_t)t * KEXT + IPAD + n] = sb;     // Ext s-part
        }
        if (blockIdx.x == 0) {                          // Ext [1 | x | pad]
            for (int c = tid; c < IPAD; c += 128)
                ext[(size_t)t * KEXT + c] = (c == 0) ? (u16)0x3F80 : x_lds[c - 1];
        }

        // device-wide step barrier
        __threadfence();
        __syncthreads();
        if (tid == 0) {
            atomicAdd(counter, 1u);
            u32 target = (u32)(t + 1) * (u32)NBLK;
            while (__hip_atomic_load(counter, __ATOMIC_ACQUIRE,
                                     __HIP_MEMORY_SCOPE_AGENT) < target)
                __builtin_amdgcn_s_sleep(2);
        }
        __syncthreads();
    }
}

// ---------------- batched readout: Out(1000x900) = Ext(1008x5024) @ W_out^T ----------------
__global__ void __launch_bounds__(128) readout_gemm(
        const u16* __restrict__ ext, const u32* __restrict__ woutP,
        float* __restrict__ outPtr) {
    const int lane  = threadIdx.x & 31;
    const int wid   = threadIdx.x >> 5;
    const int mtile = blockIdx.y * 4 + wid;
    if (mtile >= MTILES) return;
    const int ntile = blockIdx.x;
    const int m0 = mtile * 16, n0 = ntile * 16;
    const int sel = (lane & 16) ? 8 : 0;

    const u16* arow = ext + (size_t)(m0 + (lane & 15)) * KEXT;
    const u32* bb   = woutP + (size_t)ntile * NKB_OUT * 256 + (size_t)lane * 8;

    v8f acc = {};
#pragma unroll 4
    for (int kb = 0; kb < NKB_OUT; ++kb) {
        U16V a, b;
        const u16* ap = arow + kb * 32 + sel;
        a.q[0] = *(const u32x4*)(ap);
        a.q[1] = *(const u32x4*)(ap + 16);
        const u32x4* bp = (const u32x4*)(bb + (size_t)kb * 256);
        b.q[0] = bp[0];
        b.q[1] = bp[1];
        acc = __builtin_amdgcn_wmma_f32_16x16x32_bf16(
            false, a.v, false, b.v, (short)0, acc, false, false);
    }

    const int ncol = n0 + (lane & 15);
    const int mrow = m0 + ((lane & 16) ? 8 : 0);
    if (ncol < IDIM) {
#pragma unroll
        for (int r = 0; r < 8; ++r) {
            int m = mrow + r;
            if (m < SEQ) outPtr[(size_t)m * IDIM + ncol] = acc[r];
        }
    }
}

// ---------------- host launcher ----------------
extern "C" void kernel_launch(void* const* d_in, const int* in_sizes, int n_in,
                              void* d_out, int out_size, void* d_ws, size_t ws_size,
                              hipStream_t stream) {
    const float* inputs = (const float*)d_in[0];   // (1000,1,900)
    const float* state  = (const float*)d_in[1];   // (1,4096)
    const float* W_ih   = (const float*)d_in[2];   // (4096,900)
    const float* W_hh   = (const float*)d_in[3];   // (4096,4096)
    const float* W_out  = (const float*)d_in[4];   // (900,4997)

    float* outPtr    = (float*)d_out;              // outputs: 1000*900
    float* statesOut = outPtr + (size_t)SEQ * IDIM;

    char* ws = (char*)d_ws;
    u32* counter = (u32*)(ws + OFF_CNT);
    u16* sbuf0   = (u16*)(ws + OFF_SBUF0);
    u16* sbuf1   = (u16*)(ws + OFF_SBUF1);
    u16* ext     = (u16*)(ws + OFF_EXT);
    u32* wihP    = (u32*)(ws + OFF_WIH);
    u32* whhP    = (u32*)(ws + OFF_WHH);
    u32* woutP   = (u32*)(ws + OFF_WOUT);

    // 1) init
    {
        int total = HDIM + (MPAD - SEQ) * KEXT;
        esn_init<<<(total + 255) / 256, 256, 0, stream>>>(state, sbuf0, ext, counter);
    }
    // 2) pack weights to bf16 WMMA tiles (re-done each launch; deterministic)
    {
        int tot = NTILE_H * NKB_IH * 32;
        pack_bf16<<<(tot + 255) / 256, 256, 0, stream>>>(W_ih, wihP, HDIM, IDIM, IDIM, NKB_IH, 0, tot);
    }
    {
        int tot = NTILE_H * NKB_HH * 32;
        pack_bf16<<<(tot + 255) / 256, 256, 0, stream>>>(W_hh, whhP, HDIM, HDIM, HDIM, NKB_HH, 0, tot);
    }
    {
        int tot = NTILE_OUT * NKB_OUT * 32;
        pack_bf16<<<(tot + 255) / 256, 256, 0, stream>>>(W_out, woutP, IDIM, HDIM + IDIM + 1,
                                                         HDIM + IDIM + 1, NKB_OUT, 1, tot);
    }
    // 3) sequential reservoir recurrence (persistent, grid-wide barrier per step)
    esn_recurrence<<<NBLK, 128, 0, stream>>>(inputs, whhP, wihP, sbuf0, sbuf1,
                                             ext, statesOut, counter);
    // 4) batched readout GEMM over all timesteps
    readout_gemm<<<dim3(NTILE_OUT, (MTILES + 3) / 4), 128, 0, stream>>>(ext, woutP, outPtr);
}
